// TagSimple_85890755985505
// MI455X (gfx1250) — compile-verified, matrix operand order checked by reading, not run on hardware
//
#include <hip/hip_runtime.h>
#include <hip/hip_bf16.h>

typedef unsigned short u16;
typedef unsigned int u32;
typedef __bf16 bf16_t;
typedef bf16_t v16bf __attribute__((ext_vector_type(16)));
typedef float  v8f   __attribute__((ext_vector_type(8)));

#define VOCAB 50257
#define VPAD  50272   /* 3142 tiles * 16 */
#define EMB   256
#define HID   512
#define BSZ   64
#define SEQ   512
#define KCAT  768     /* EMB + HID */
#define NGATE 2048    /* 4*HID */

union BF16x16 { v16bf v; uint4 q[2]; };

__device__ __forceinline__ u16 f2bf(float x) {
    u32 u = __float_as_uint(x);
    u32 r = u + 0x7FFFu + ((u >> 16) & 1u);
    return (u16)(r >> 16);
}
__device__ __forceinline__ float sigm(float x) { return 1.0f / (1.0f + __expf(-x)); }

// Load a 16-wide bf16 operand fragment for wmma (A row-major or B from WT[N][K]):
// lanes 0-15 hold K {k0..k0+7, k0+16..k0+23}; lanes 16-31 hold K {k0+8.., k0+24..}
__device__ __forceinline__ v16bf load_frag(const u16* rowbase, int k0, bool hi) {
    BF16x16 u;
    int e0 = k0 + (hi ? 8 : 0);
    u.q[0] = *reinterpret_cast<const uint4*>(rowbase + e0);
    u.q[1] = *reinterpret_cast<const uint4*>(rowbase + e0 + 16);
    return u.v;
}

// ---------------- init / conversion kernels ----------------

__global__ void k_zero_state(u16* h0, float* c) {
    int i = blockIdx.x * blockDim.x + threadIdx.x;
    if (i < BSZ * HID) { h0[i] = 0; c[i] = 0.0f; }
}

__global__ void k_gather_emb(const int* __restrict__ inputs, const float* __restrict__ emb,
                             u16* __restrict__ Xbf) {
    int i = blockIdx.x * blockDim.x + threadIdx.x;  // over B*S*EMB
    if (i >= BSZ * SEQ * EMB) return;
    int row = i / EMB, e = i % EMB;
    int tok = inputs[row];
    Xbf[i] = f2bf(emb[(size_t)tok * EMB + e]);
}

// WcatT[n*768 + k] = k<256 ? Wi[k][n] : Wh[k-256][n]   (n in [0,2048))
__global__ void k_conv_wcat(const float* __restrict__ Wi, const float* __restrict__ Wh,
                            u16* __restrict__ WT) {
    int i = blockIdx.x * blockDim.x + threadIdx.x;  // over 2048*768
    if (i >= NGATE * KCAT) return;
    int n = i / KCAT, k = i % KCAT;
    float v = (k < EMB) ? Wi[(size_t)k * NGATE + n] : Wh[(size_t)(k - EMB) * NGATE + n];
    WT[i] = f2bf(v);
}

// square 512x512 transpose to bf16: WT[n*512+k] = W[k*512+n]
__global__ void k_conv_sq(const float* __restrict__ W, u16* __restrict__ WT) {
    int i = blockIdx.x * blockDim.x + threadIdx.x;
    if (i >= HID * HID) return;
    int n = i / HID, k = i % HID;
    WT[i] = f2bf(W[(size_t)k * HID + n]);
}

// W3T [VPAD][512], zero padded rows
__global__ void k_conv_w3(const float* __restrict__ W3, u16* __restrict__ WT) {
    int i = blockIdx.x * blockDim.x + threadIdx.x;  // over VPAD*512
    if (i >= VPAD * HID) return;
    int n = i / HID, k = i % HID;
    WT[i] = (n < VOCAB) ? f2bf(W3[(size_t)k * VOCAB + n]) : (u16)0;
}

// ---------------- LSTM step: z = [x_t||h] @ WcatT + b ; gates ; update c,h ----------------
// grid (32 hidden-tiles j, 4 m-tiles), block 128 = 4 waves, wave w = gate g.
__global__ void k_lstm_step(const u16* __restrict__ Xbf, const u16* __restrict__ WT,
                            const float* __restrict__ bias, const u16* __restrict__ h_in,
                            u16* __restrict__ h_out, float* __restrict__ c, int t) {
    __shared__ float zs[4][16][16];  // [gate][M][N]

    const int tid  = threadIdx.x;
    const int wave = tid >> 5;         // gate index 0..3 (i,f,g,o)
    const int lane = tid & 31;
    const int lrow = lane & 15;
    const bool hi  = lane >= 16;
    const int j    = blockIdx.x;       // hidden tile 0..31
    const int mt   = blockIdx.y;       // m tile 0..3
    const int m0   = mt * 16;
    const int n0   = wave * HID + j * 16;   // column in z (gate block)

    const int am = m0 + lrow;                                  // A row (batch)
    const u16* xrow = Xbf + ((size_t)am * SEQ + t) * EMB;
    const u16* hrow = h_in + (size_t)am * HID;
    const u16* wrow = WT + (size_t)(n0 + lrow) * KCAT;

    v8f acc = {};
    #pragma unroll 4
    for (int kc = 0; kc < 8; ++kc) {   // x part, K 0..255
        v16bf a = load_frag(xrow, kc * 32, hi);
        v16bf b = load_frag(wrow, kc * 32, hi);
        acc = __builtin_amdgcn_wmma_f32_16x16x32_bf16(false, a, false, b, (short)0, acc, false, false);
    }
    #pragma unroll 4
    for (int kc = 8; kc < 24; ++kc) {  // h part, K 256..767
        v16bf a = load_frag(hrow, kc * 32 - EMB, hi);
        v16bf b = load_frag(wrow, kc * 32, hi);
        acc = __builtin_amdgcn_wmma_f32_16x16x32_bf16(false, a, false, b, (short)0, acc, false, false);
    }

    float bn = bias[n0 + lrow];
    #pragma unroll
    for (int r = 0; r < 8; ++r) {
        int mm = r + (hi ? 8 : 0);
        zs[wave][mm][lrow] = acc[r] + bn;
    }
    __syncthreads();

    // gate math: 256 elements, 128 threads -> 2 each
    for (int idx = tid; idx < 256; idx += 128) {
        int mm = idx >> 4, nn = idx & 15;
        float zi = zs[0][mm][nn], zf = zs[1][mm][nn];
        float zg = zs[2][mm][nn], zo = zs[3][mm][nn];
        int row = m0 + mm;
        int u   = j * 16 + nn;
        size_t ci = (size_t)row * HID + u;
        float cn = sigm(zf) * c[ci] + sigm(zi) * tanhf(zg);
        float hn = sigm(zo) * tanhf(cn);
        c[ci] = cn;
        h_out[ci] = f2bf(hn);
    }
}

// ---------------- dense: out = tanh(A @ WT^T + b), bf16 out. block 128 = 4 waves = 4 m-tiles ----------------
__global__ void k_gemm_tanh(const u16* __restrict__ A, const u16* __restrict__ WT,
                            const float* __restrict__ bias, u16* __restrict__ out, int K, int N) {
    const int tid = threadIdx.x;
    const int mt  = tid >> 5;
    const int lane = tid & 31;
    const int lrow = lane & 15;
    const bool hi  = lane >= 16;
    const int n0 = blockIdx.x * 16;
    const int m0 = mt * 16;

    const u16* arow = A  + (size_t)(m0 + lrow) * K;
    const u16* wrow = WT + (size_t)(n0 + lrow) * K;

    v8f acc = {};
    for (int kc = 0; kc < K / 32; ++kc) {
        v16bf a = load_frag(arow, kc * 32, hi);
        v16bf b = load_frag(wrow, kc * 32, hi);
        acc = __builtin_amdgcn_wmma_f32_16x16x32_bf16(false, a, false, b, (short)0, acc, false, false);
    }
    float bn = bias[n0 + lrow];
    #pragma unroll
    for (int r = 0; r < 8; ++r) {
        int m = m0 + r + (hi ? 8 : 0);
        out[(size_t)m * N + n0 + lrow] = f2bf(tanhf(acc[r] + bn));
    }
}

// ---------------- logits: fp32 out with bias, N bounded (padded WT rows are zero) ----------------
__global__ void k_gemm_logits(const u16* __restrict__ A, const u16* __restrict__ WT,
                              const float* __restrict__ bias, float* __restrict__ out, int K, int N) {
    const int tid = threadIdx.x;
    const int mt  = tid >> 5;
    const int lane = tid & 31;
    const int lrow = lane & 15;
    const bool hi  = lane >= 16;
    const int n0 = blockIdx.x * 16;
    const int m0 = mt * 16;

    const u16* arow = A  + (size_t)(m0 + lrow) * K;
    const u16* wrow = WT + (size_t)(n0 + lrow) * K;

    v8f acc = {};
    #pragma unroll 4
    for (int kc = 0; kc < 16; ++kc) {  // K = 512
        v16bf a = load_frag(arow, kc * 32, hi);
        v16bf b = load_frag(wrow, kc * 32, hi);
        acc = __builtin_amdgcn_wmma_f32_16x16x32_bf16(false, a, false, b, (short)0, acc, false, false);
    }
    int n = n0 + lrow;
    if (n < N) {
        float bn = bias[n];
        #pragma unroll
        for (int r = 0; r < 8; ++r) {
            int m = m0 + r + (hi ? 8 : 0);
            out[(size_t)m * N + n] = acc[r] + bn;
        }
    }
}

// ---------------- row-wise log_softmax ----------------
__global__ void k_log_softmax(const float* __restrict__ logits, float* __restrict__ out, int N) {
    __shared__ float red[256];
    const int row = blockIdx.x;
    const float* x = logits + (size_t)row * N;
    float* o = out + (size_t)row * N;

    float mx = -3.4e38f;
    for (int i = threadIdx.x; i < N; i += 256) mx = fmaxf(mx, x[i]);
    red[threadIdx.x] = mx; __syncthreads();
    for (int s = 128; s > 0; s >>= 1) {
        if (threadIdx.x < s) red[threadIdx.x] = fmaxf(red[threadIdx.x], red[threadIdx.x + s]);
        __syncthreads();
    }
    mx = red[0]; __syncthreads();

    float sum = 0.0f;
    for (int i = threadIdx.x; i < N; i += 256) sum += expf(x[i] - mx);
    red[threadIdx.x] = sum; __syncthreads();
    for (int s = 128; s > 0; s >>= 1) {
        if (threadIdx.x < s) red[threadIdx.x] += red[threadIdx.x + s];
        __syncthreads();
    }
    float lse = mx + logf(red[0]);
    for (int i = threadIdx.x; i < N; i += 256) o[i] = x[i] - lse;
}

// ---------------- host orchestration ----------------

static inline size_t alignup(size_t x) { return (x + 255) & ~(size_t)255; }

extern "C" void kernel_launch(void* const* d_in, const int* in_sizes, int n_in,
                              void* d_out, int out_size, void* d_ws, size_t ws_size,
                              hipStream_t stream) {
    const int*   inputs = (const int*)  d_in[0];
    const float* emb    = (const float*)d_in[1];
    const float* Wi     = (const float*)d_in[2];
    const float* Wh     = (const float*)d_in[3];
    const float* b      = (const float*)d_in[4];
    const float* W1     = (const float*)d_in[5];
    const float* b1     = (const float*)d_in[6];
    const float* W2     = (const float*)d_in[7];
    const float* b2     = (const float*)d_in[8];
    const float* W3     = (const float*)d_in[9];
    const float* b3     = (const float*)d_in[10];

    char* ws = (char*)d_ws;
    size_t off = 0;
    u16*   Xbf    = (u16*)(ws + off); off = alignup(off + (size_t)BSZ * SEQ * EMB * 2);
    u16*   WcatT  = (u16*)(ws + off); off = alignup(off + (size_t)NGATE * KCAT * 2);
    u16*   W1T    = (u16*)(ws + off); off = alignup(off + (size_t)HID * HID * 2);
    u16*   W2T    = (u16*)(ws + off); off = alignup(off + (size_t)HID * HID * 2);
    u16*   W3T    = (u16*)(ws + off); off = alignup(off + (size_t)VPAD * HID * 2);
    u16*   h0     = (u16*)(ws + off); off = alignup(off + (size_t)BSZ * HID * 2);
    u16*   h1     = (u16*)(ws + off); off = alignup(off + (size_t)BSZ * HID * 2);
    float* cbuf   = (float*)(ws + off); off = alignup(off + (size_t)BSZ * HID * 4);
    u16*   y1     = (u16*)(ws + off); off = alignup(off + (size_t)BSZ * HID * 2);
    u16*   y2     = (u16*)(ws + off); off = alignup(off + (size_t)BSZ * HID * 2);
    float* logits = (float*)(ws + off); off = alignup(off + (size_t)BSZ * VOCAB * 4);
    (void)ws_size; (void)in_sizes; (void)n_in; (void)out_size;

    // init + weight conversion (all on stream, serialized)
    k_zero_state<<<(BSZ * HID + 255) / 256, 256, 0, stream>>>(h0, cbuf);
    k_gather_emb<<<(BSZ * SEQ * EMB + 255) / 256, 256, 0, stream>>>(inputs, emb, Xbf);
    k_conv_wcat<<<(NGATE * KCAT + 255) / 256, 256, 0, stream>>>(Wi, Wh, WcatT);
    k_conv_sq<<<(HID * HID + 255) / 256, 256, 0, stream>>>(W1, W1T);
    k_conv_sq<<<(HID * HID + 255) / 256, 256, 0, stream>>>(W2, W2T);
    k_conv_w3<<<(VPAD * HID + 255) / 256, 256, 0, stream>>>(W3, W3T);

    // sequential LSTM recurrence, double-buffered h
    dim3 sgrid(HID / 16, 4);
    for (int t = 0; t < SEQ; ++t) {
        const u16* hin = (t & 1) ? h1 : h0;
        u16*       hout = (t & 1) ? h0 : h1;
        k_lstm_step<<<sgrid, 128, 0, stream>>>(Xbf, WcatT, b, hin, hout, cbuf, t);
    }
    // after t=511 (odd), final h lives in h0

    k_gemm_tanh<<<HID / 16, 128, 0, stream>>>(h0, W1T, b1, y1, HID, HID);
    k_gemm_tanh<<<HID / 16, 128, 0, stream>>>(y1, W2T, b2, y2, HID, HID);
    k_gemm_logits<<<VPAD / 16, 128, 0, stream>>>(y2, W3T, b3, logits, HID, VOCAB);
    k_log_softmax<<<BSZ, 256, 0, stream>>>(logits, (float*)d_out, VOCAB);
}
